// ASPLayer_472446402583
// MI455X (gfx1250) — matrix operand order repaired and checked by reading, not compiled, not run on hardware
//
#include <hip/hip_runtime.h>

// Problem constants (B, N, D, R) from the reference.
#define BB 4
#define NN 2048
#define DD 1024
#define RR 64
#define LN_EPS 1e-5f

typedef __attribute__((ext_vector_type(16))) __bf16 v16bf;
typedef __attribute__((ext_vector_type(2)))  __bf16 v2bf;
typedef __attribute__((ext_vector_type(8)))  float  v8f;
typedef __attribute__((ext_vector_type(4)))  float  v4f;
typedef __attribute__((ext_vector_type(8)))  unsigned short su8;

union BF16Frag {
    v16bf v;
    su8 h[2];
    unsigned u[8];
    unsigned short s[16];
};

// Pack two fp32 into packed bf16 pair (backend selects v_cvt_pk_bf16_f32).
__device__ inline unsigned pk_bf16(float a, float b) {
    union { v2bf v; unsigned u; } c;
    c.v[0] = (__bf16)a;
    c.v[1] = (__bf16)b;
    return c.u;
}

__device__ inline unsigned short bf16bits(float f) {
    union { __bf16 h; unsigned short s; } c;
    c.h = (__bf16)f;
    return c.s;
}

__device__ inline v8f zero8() {
    v8f z;
    #pragma unroll
    for (int i = 0; i < 8; ++i) z[i] = 0.0f;
    return z;
}

__device__ inline v8f wmma_bf16(const BF16Frag& a, const BF16Frag& b, v8f c) {
    return __builtin_amdgcn_wmma_f32_16x16x32_bf16(
        /*neg_a=*/false, a.v, /*neg_b=*/false, b.v,
        /*c_mod=*/(short)0, c, /*reuse_a=*/false, /*reuse_b=*/false);
}

// ---------------------------------------------------------------------------
// Kernel 0: pack U, V (D x R fp32) into U^T, V^T (R x D bf16, row-major).
// Rows of U^T (fixed r, contiguous d) are exactly the WMMA B-operand layout.
// ---------------------------------------------------------------------------
__global__ __launch_bounds__(256) void pack_uv(const float* __restrict__ U,
                                               const float* __restrict__ V,
                                               unsigned short* __restrict__ Ut,
                                               unsigned short* __restrict__ Vt) {
    int idx = blockIdx.x * 256 + threadIdx.x;      // R*D/2 = 32768 pairs
    if (idx >= RR * DD / 2) return;
    int r  = idx >> 9;             // / (D/2)
    int d2 = (idx & (DD / 2 - 1)) * 2;
    unsigned up = pk_bf16(U[(size_t)d2 * RR + r], U[(size_t)(d2 + 1) * RR + r]);
    unsigned vp = pk_bf16(V[(size_t)d2 * RR + r], V[(size_t)(d2 + 1) * RR + r]);
    *(unsigned*)&Ut[(size_t)r * DD + d2] = up;
    *(unsigned*)&Vt[(size_t)r * DD + d2] = vp;
}

// ---------------------------------------------------------------------------
// Kernel 1: transpose x (B,N,D fp32) -> xT (B,D,N bf16) via LDS 64x64 tiles.
// ---------------------------------------------------------------------------
__global__ __launch_bounds__(256) void xpose(const float* __restrict__ x,
                                             unsigned short* __restrict__ xT) {
    __shared__ float tile[64][65];
    const int dTiles = DD / 64;   // 16
    const int nTiles = NN / 64;   // 32
    int bid = blockIdx.x;
    int b   = bid / (nTiles * dTiles);
    int rem = bid % (nTiles * dTiles);
    int nt  = rem / dTiles, dt = rem % dTiles;
    int n0 = nt * 64, d0 = dt * 64;
    const float* xb = x + (size_t)b * NN * DD;
    unsigned short* xTb = xT + (size_t)b * DD * NN;
    #pragma unroll
    for (int k = 0; k < 16; ++k) {
        int idx = k * 256 + threadIdx.x;
        int r = idx >> 6, c = idx & 63;
        tile[r][c] = xb[(size_t)(n0 + r) * DD + d0 + c];
    }
    __syncthreads();
    #pragma unroll
    for (int k = 0; k < 8; ++k) {
        int idx = k * 256 + threadIdx.x;      // 2048 pair slots
        int dr = idx >> 5;
        int nc = (idx & 31) * 2;
        unsigned p = pk_bf16(tile[nc][dr], tile[nc + 1][dr]);
        *(unsigned*)&xTb[(size_t)(d0 + dr) * NN + n0 + nc] = p;
    }
}

// ---------------------------------------------------------------------------
// Kernel 2: Q' = (x@U) * mask / sqrt(max(sum(mask),1)),  K' = (x@V) * mask.
// 256 threads = 8 waves; each wave owns 16 tokens; contraction over D in
// K=32 chunks, 8 WMMA per chunk per wave (4 col-tiles x {Q,K}).
// ---------------------------------------------------------------------------
__global__ __launch_bounds__(256) void proj_qk(const float* __restrict__ x,
                                               const float* __restrict__ mask,
                                               const unsigned short* __restrict__ Ut,
                                               const unsigned short* __restrict__ Vt,
                                               unsigned short* __restrict__ Qs,
                                               unsigned short* __restrict__ Ks) {
    int tid = threadIdx.x, wave = tid >> 5, lane = tid & 31;
    int hi = lane >> 4, ln = lane & 15;
    int q0 = blockIdx.x * 128 + wave * 16;         // global token base of this wave

    v8f accQ[4], accK[4];
    #pragma unroll
    for (int t = 0; t < 4; ++t) { accQ[t] = zero8(); accK[t] = zero8(); }

    const float* xrow = x + (size_t)(q0 + ln) * DD;

    for (int c = 0; c < DD / 32; ++c) {
        int db = c * 32 + (hi ? 8 : 0);
        v4f f0 = *(const v4f*)(xrow + db);
        v4f f1 = *(const v4f*)(xrow + db + 4);
        v4f f2 = *(const v4f*)(xrow + db + 16);
        v4f f3 = *(const v4f*)(xrow + db + 20);
        BF16Frag a;
        a.u[0] = pk_bf16(f0[0], f0[1]);
        a.u[1] = pk_bf16(f0[2], f0[3]);
        a.u[2] = pk_bf16(f1[0], f1[1]);
        a.u[3] = pk_bf16(f1[2], f1[3]);
        a.u[4] = pk_bf16(f2[0], f2[1]);
        a.u[5] = pk_bf16(f2[2], f2[3]);
        a.u[6] = pk_bf16(f3[0], f3[1]);
        a.u[7] = pk_bf16(f3[2], f3[3]);
        int rb = c * 32 + (hi ? 8 : 0);
        #pragma unroll
        for (int t = 0; t < 4; ++t) {
            BF16Frag bu, bv;
            const unsigned short* up = Ut + (size_t)(t * 16 + ln) * DD + rb;
            const unsigned short* vp = Vt + (size_t)(t * 16 + ln) * DD + rb;
            bu.h[0] = *(const su8*)up; bu.h[1] = *(const su8*)(up + 16);
            bv.h[0] = *(const su8*)vp; bv.h[1] = *(const su8*)(vp + 16);
            accQ[t] = wmma_bf16(a, bu, accQ[t]);
            accK[t] = wmma_bf16(a, bv, accK[t]);
        }
    }

    float mv[4][8];
    float msum[8];
    #pragma unroll
    for (int i = 0; i < 8; ++i) msum[i] = 0.0f;
    #pragma unroll
    for (int t = 0; t < 4; ++t) {
        #pragma unroll
        for (int i = 0; i < 8; ++i) {
            int row = q0 + i + hi * 8;
            float m = mask[(size_t)row * RR + t * 16 + ln];
            mv[t][i] = m;
            msum[i] += m;
        }
    }
    float qsc[8];
    #pragma unroll
    for (int i = 0; i < 8; ++i) {
        float s = msum[i];
        #pragma unroll
        for (int off = 1; off < 16; off <<= 1) s += __shfl_xor(s, off, 32);
        qsc[i] = rsqrtf(fmaxf(s, 1.0f));
    }
    #pragma unroll
    for (int t = 0; t < 4; ++t) {
        #pragma unroll
        for (int i = 0; i < 8; ++i) {
            int row = q0 + i + hi * 8;
            size_t idx = (size_t)row * RR + t * 16 + ln;
            Qs[idx] = bf16bits(accQ[t][i] * mv[t][i] * qsc[i]);
            Ks[idx] = bf16bits(accK[t][i] * mv[t][i]);
        }
    }
}

// ---------------------------------------------------------------------------
// Kernel 3: flash attention + residual LayerNorm.
// Block = 16 queries, 128 threads = 4 waves; each wave owns a 256-wide D
// slice (16 col-tiles, 128 acc VGPRs). Per 64-key step per wave:
//   8 WMMA  S = Q'K'^T   (redundant across only 4 waves)
//   2 WMMA  l += P @ ones (row-sum via matrix core, no shuffle reduction)
//  32 WMMA  delta += P @ xT  (unique work, ~76% of matrix issue)
// Online softmax row-max uses shfl_xor within the 16-lane N-groups of the
// C-layout; P converts C-layout -> A-layout through per-wave LDS (same-wave
// DS ops are in-order, no barrier).
// ---------------------------------------------------------------------------
__global__ __launch_bounds__(128) void attn(const float* __restrict__ x,
                                            const unsigned short* __restrict__ Qs,
                                            const unsigned short* __restrict__ Ks,
                                            const unsigned short* __restrict__ xT,
                                            const float* __restrict__ gamma,
                                            const float* __restrict__ beta,
                                            float* __restrict__ out) {
    __shared__ __align__(16) unsigned short Pl[4][16][64];
    __shared__ float redS[4][16], redQ[4][16];
    __shared__ float mu_s[16], inv_s[16];

    int tid = threadIdx.x, wave = tid >> 5, lane = tid & 31;
    int hi = lane >> 4, ln = lane & 15;
    int g0 = blockIdx.x * 16;          // global query token base
    int b  = g0 / NN;
    int kb_tok = b * NN;
    const unsigned short* xTb = xT + (size_t)b * DD * NN;

    // Q' A-operand fragments (persist the whole k-loop).
    BF16Frag qa[2];
    {
        const unsigned short* qrow = Qs + (size_t)(g0 + ln) * RR;
        #pragma unroll
        for (int c = 0; c < 2; ++c) {
            const unsigned short* p = qrow + c * 32 + (hi ? 8 : 0);
            qa[c].h[0] = *(const su8*)p;
            qa[c].h[1] = *(const su8*)(p + 16);
        }
    }
    // All-ones B fragment: row-sum of P via WMMA.
    BF16Frag ones;
    #pragma unroll
    for (int j = 0; j < 8; ++j) ones.u[j] = 0x3F803F80u;

    v8f acc[16];
    #pragma unroll
    for (int ct = 0; ct < 16; ++ct) acc[ct] = zero8();
    v8f acc_l = zero8();
    float m[8];
    #pragma unroll
    for (int i = 0; i < 8; ++i) m[i] = -3.0e38f;

    for (int k0 = 0; k0 < NN; k0 += 64) {
        if (k0 + 64 < NN) {
            __builtin_prefetch(Ks + (size_t)(kb_tok + k0 + 64 + lane) * RR, 0, 0);
            __builtin_prefetch(Ks + (size_t)(kb_tok + k0 + 96 + lane) * RR, 0, 0);
            __builtin_prefetch(xTb + (size_t)(wave * 256 + lane * 8) * NN + k0 + 64, 0, 0);
        }

        // K' B-operands: four 16-key tiles x two K=32 chunks of R.
        BF16Frag kbf[4][2];
        #pragma unroll
        for (int j = 0; j < 4; ++j) {
            const unsigned short* kr = Ks + (size_t)(kb_tok + k0 + 16 * j + ln) * RR;
            #pragma unroll
            for (int c = 0; c < 2; ++c) {
                const unsigned short* p = kr + c * 32 + (hi ? 8 : 0);
                kbf[j][c].h[0] = *(const su8*)p;
                kbf[j][c].h[1] = *(const su8*)(p + 16);
            }
        }
        v8f S[4];
        #pragma unroll
        for (int j = 0; j < 4; ++j) {
            S[j] = zero8();
            S[j] = wmma_bf16(qa[0], kbf[j][0], S[j]);
            S[j] = wmma_bf16(qa[1], kbf[j][1], S[j]);
        }

        // Online softmax over this 64-key slab.
        float alpha[8];
        #pragma unroll
        for (int i = 0; i < 8; ++i) {
            float r = fmaxf(fmaxf(S[0][i], S[1][i]), fmaxf(S[2][i], S[3][i]));
            #pragma unroll
            for (int off = 1; off < 16; off <<= 1) r = fmaxf(r, __shfl_xor(r, off, 32));
            float mn = fmaxf(m[i], r);
            alpha[i] = __expf(m[i] - mn);
            m[i] = mn;
            #pragma unroll
            for (int j = 0; j < 4; ++j) S[j][i] = __expf(S[j][i] - mn);
        }

        // P: C-layout -> LDS (packed bf16 pairs) -> A-layout fragments.
        #pragma unroll
        for (int i = 0; i < 8; ++i) {
            int row = i + hi * 8;
            unsigned p01 = pk_bf16(S[0][i], S[1][i]);
            unsigned p23 = pk_bf16(S[2][i], S[3][i]);
            Pl[wave][row][ln]      = (unsigned short)p01;
            Pl[wave][row][16 + ln] = (unsigned short)(p01 >> 16);
            Pl[wave][row][32 + ln] = (unsigned short)p23;
            Pl[wave][row][48 + ln] = (unsigned short)(p23 >> 16);
        }
        BF16Frag pa[2];
        #pragma unroll
        for (int c = 0; c < 2; ++c) {
            const unsigned short* pp = &Pl[wave][ln][c * 32 + (hi ? 8 : 0)];
            pa[c].h[0] = *(const su8*)pp;
            pa[c].h[1] = *(const su8*)(pp + 16);
        }

        // Rescale running state by alpha (per-row, matching C-layout mapping).
        #pragma unroll
        for (int i = 0; i < 8; ++i) acc_l[i] *= alpha[i];
        #pragma unroll
        for (int ct = 0; ct < 16; ++ct) {
            #pragma unroll
            for (int i = 0; i < 8; ++i) acc[ct][i] *= alpha[i];
        }
        // l += rowsum(P) via P @ ones.
        acc_l = wmma_bf16(pa[0], ones, acc_l);
        acc_l = wmma_bf16(pa[1], ones, acc_l);

        // delta accumulation over this wave's 256-wide D slice.
        #pragma unroll
        for (int ct = 0; ct < 16; ++ct) {
            #pragma unroll
            for (int c = 0; c < 2; ++c) {
                BF16Frag xb;
                const unsigned short* xp =
                    xTb + (size_t)(wave * 256 + ct * 16 + ln) * NN + k0 + c * 32 + (hi ? 8 : 0);
                xb.h[0] = *(const su8*)xp;
                xb.h[1] = *(const su8*)(xp + 16);
                acc[ct] = wmma_bf16(pa[c], xb, acc[ct]);
            }
        }
    }

    // y = x + delta; LayerNorm across D (cross-wave reduction through LDS).
    float linv[8];
    #pragma unroll
    for (int i = 0; i < 8; ++i) linv[i] = 1.0f / acc_l[i];

    #pragma unroll
    for (int ct = 0; ct < 16; ++ct) {
        #pragma unroll
        for (int i = 0; i < 8; ++i) {
            int row = g0 + i + hi * 8;
            int d = wave * 256 + ct * 16 + ln;
            float y = x[(size_t)row * DD + d] + acc[ct][i] * linv[i];
            acc[ct][i] = y;
        }
    }
    #pragma unroll
    for (int i = 0; i < 8; ++i) {
        float s = 0.0f, q = 0.0f;
        #pragma unroll
        for (int ct = 0; ct < 16; ++ct) { s += acc[ct][i]; q += acc[ct][i] * acc[ct][i]; }
        #pragma unroll
        for (int off = 1; off < 16; off <<= 1) {
            s += __shfl_xor(s, off, 32);
            q += __shfl_xor(q, off, 32);
        }
        if (ln == 0) {
            redS[wave][i + hi * 8] = s;
            redQ[wave][i + hi * 8] = q;
        }
    }
    __syncthreads();
    if (tid < 16) {
        float s = 0.0f, q = 0.0f;
        #pragma unroll
        for (int w = 0; w < 4; ++w) { s += redS[w][tid]; q += redQ[w][tid]; }
        float mu  = s * (1.0f / DD);
        float var = q * (1.0f / DD) - mu * mu;
        mu_s[tid]  = mu;
        inv_s[tid] = rsqrtf(var + LN_EPS);
    }
    __syncthreads();
    #pragma unroll
    for (int ct = 0; ct < 16; ++ct) {
        int d = wave * 256 + ct * 16 + ln;
        float g = gamma[d], be = beta[d];
        #pragma unroll
        for (int i = 0; i < 8; ++i) {
            int row = g0 + i + hi * 8;
            int rloc = i + hi * 8;
            out[(size_t)row * DD + d] = (acc[ct][i] - mu_s[rloc]) * inv_s[rloc] * g + be;
        }
    }
}

// ---------------------------------------------------------------------------
extern "C" void kernel_launch(void* const* d_in, const int* in_sizes, int n_in,
                              void* d_out, int out_size, void* d_ws, size_t ws_size,
                              hipStream_t stream) {
    const float* x     = (const float*)d_in[0];
    const float* mask  = (const float*)d_in[1];
    const float* U     = (const float*)d_in[2];
    const float* V     = (const float*)d_in[3];
    const float* gamma = (const float*)d_in[4];
    const float* beta  = (const float*)d_in[5];
    float* out = (float*)d_out;

    char* w = (char*)d_ws;
    size_t off = 0;
    auto wsalloc = [&](size_t bytes) -> void* {
        void* p = w + off;
        off += (bytes + 255) & ~(size_t)255;
        return p;
    };
    unsigned short* xT = (unsigned short*)wsalloc((size_t)BB * DD * NN * 2);
    unsigned short* Qs = (unsigned short*)wsalloc((size_t)BB * NN * RR * 2);
    unsigned short* Ks = (unsigned short*)wsalloc((size_t)BB * NN * RR * 2);
    unsigned short* Ut = (unsigned short*)wsalloc((size_t)RR * DD * 2);
    unsigned short* Vt = (unsigned short*)wsalloc((size_t)RR * DD * 2);

    pack_uv<<<(RR * DD / 2 + 255) / 256, 256, 0, stream>>>(U, V, Ut, Vt);
    xpose<<<BB * (NN / 64) * (DD / 64), 256, 0, stream>>>(x, xT);
    proj_qk<<<(BB * NN) / 128, 256, 0, stream>>>(x, mask, Ut, Vt, Qs, Ks);
    attn<<<(BB * NN) / 16, 128, 0, stream>>>(x, Qs, Ks, xT, gamma, beta, out);
}
